// MultiHeadAttentionEfficient_85761906967093
// MI455X (gfx1250) — matrix operand order repaired and checked by reading, not compiled, not run on hardware
//
#include <hip/hip_runtime.h>
#include <hip/hip_fp16.h>

typedef __attribute__((ext_vector_type(16))) _Float16 v16h;
typedef __attribute__((ext_vector_type(8)))  _Float16 v8h;
typedef __attribute__((ext_vector_type(8)))  float    v8f;

#define N_  2
#define T_  2048
#define D_  1024
#define H_  16
#define DH_ 64
#define TD3 (3*D_)

__device__ __forceinline__ v16h cat8(v8h a, v8h b) {
    v16h r;
#pragma unroll
    for (int i = 0; i < 8; ++i) { r[i] = a[i]; r[i + 8] = b[i]; }
    return r;
}

// ---------------------------------------------------------------------------
// Kernel 0: fp32 -> f16 conversion (grid-stride)
// ---------------------------------------------------------------------------
__global__ void cvt_f32_to_f16(const float* __restrict__ src,
                               _Float16* __restrict__ dst, int count) {
    int i = blockIdx.x * blockDim.x + threadIdx.x;
    int stride = gridDim.x * blockDim.x;
    for (; i < count; i += stride) dst[i] = (_Float16)src[i];
}

// ---------------------------------------------------------------------------
// Kernel 1: QKV GEMM  [4096,1024] x [1024,3072] -> q/k/v [N,H,T,DH] (f16)
// Block tile 128(M) x 64(N), K-steps of 32, 8 waves: 4x2 wave grid, each wave
// does a 32x32 sub-tile = 2x2 WMMA tiles. q is pre-scaled by DH^-0.5.
// ---------------------------------------------------------------------------
__global__ __launch_bounds__(256) void qkv_gemm(
        const _Float16* __restrict__ A,   // xh  [4096][1024]
        const _Float16* __restrict__ B,   // Wh  [1024][3072]
        _Float16* __restrict__ qh, _Float16* __restrict__ kh,
        _Float16* __restrict__ vh) {
    __shared__ __align__(32) _Float16 As[128 * 32];  // [m][k]
    __shared__ __align__(32) _Float16 Bs[64 * 32];   // Bt[n][k]
    const int Ntiles = TD3 / 64;                     // 48
    int mblock = (blockIdx.x / Ntiles) * 128;
    int cblock = (blockIdx.x % Ntiles) * 64;
    int tid = threadIdx.x;
    int lane = tid & 31, w = tid >> 5;
    int wrow = w & 3, wcol = w >> 2;
    int nlo = lane & 15, hi = lane >> 4;

    v8f acc[2][2];
#pragma unroll
    for (int i = 0; i < 2; ++i)
#pragma unroll
        for (int j = 0; j < 2; ++j) { v8f z = {}; acc[i][j] = z; }

    for (int kb = 0; kb < D_; kb += 32) {
        __syncthreads();
        {   // stage A 128x32: thread -> (row, 16-half chunk)
            int r = tid >> 1, c = (tid & 1) * 16;
            const v8h* g = (const v8h*)(A + (size_t)(mblock + r) * D_ + kb + c);
            *(v8h*)(&As[r * 32 + c])     = g[0];
            *(v8h*)(&As[r * 32 + c + 8]) = g[1];
        }
        {   // stage Bt[n][k] = W[kb+k][cblock+n] (transpose in staging)
            int c = tid & 63, k0 = (tid >> 6) * 8;
#pragma unroll
            for (int i = 0; i < 8; ++i)
                Bs[c * 32 + k0 + i] = B[(size_t)(kb + k0 + i) * TD3 + cblock + c];
        }
        __syncthreads();

        v16h aop[2], bop[2];
#pragma unroll
        for (int i = 0; i < 2; ++i) {
            int m = wrow * 32 + i * 16 + nlo;      // A-layout: lane row = L%16
            v8h c0 = *(const v8h*)(&As[m * 32 + hi * 16]);
            v8h c1 = *(const v8h*)(&As[m * 32 + hi * 16 + 8]);
            aop[i] = cat8(c0, c1);
        }
#pragma unroll
        for (int j = 0; j < 2; ++j) {
            int nn = wcol * 32 + j * 16 + nlo;     // B-layout: lane col = L%16
            bop[j] = *(const v16h*)(&Bs[nn * 32 + hi * 16]);
        }
#pragma unroll
        for (int i = 0; i < 2; ++i)
#pragma unroll
            for (int j = 0; j < 2; ++j)
                acc[i][j] = __builtin_amdgcn_wmma_f32_16x16x32_f16(
                    false, aop[i], false, bop[j], (short)0, acc[i][j], false, false);
    }

    const float qscale = 0.125f;  // DH^-0.5
#pragma unroll
    for (int i = 0; i < 2; ++i)
#pragma unroll
        for (int j = 0; j < 2; ++j)
#pragma unroll
            for (int r = 0; r < 8; ++r) {
                int Rg = mblock + wrow * 32 + i * 16 + r + 8 * hi;
                int Cg = cblock + wcol * 32 + j * 16 + nlo;
                float val = acc[i][j][r];
                int part = Cg >> 10, within = Cg & 1023;
                int hd = within >> 6, dh = within & 63;
                int nI = Rg >> 11, tI = Rg & (T_ - 1);
                size_t off = ((size_t)(nI * H_ + hd) * T_ + tI) * DH_ + dh;
                if (part == 0)      qh[off] = (_Float16)(val * qscale);
                else if (part == 1) kh[off] = (_Float16)val;
                else                vh[off] = (_Float16)val;
            }
}

// ---------------------------------------------------------------------------
// Kernel 2: flash attention. One workgroup = (n, h, 128 query rows).
// Each of 8 waves owns a 16-row query strip; K/V 64x64 tiles staged in LDS.
// ---------------------------------------------------------------------------
__global__ __launch_bounds__(256) void attn(
        const _Float16* __restrict__ qh, const _Float16* __restrict__ kh,
        const _Float16* __restrict__ vh, _Float16* __restrict__ atth) {
    __shared__ __align__(32) _Float16 Kl[64 * 64];       // [s][d]
    __shared__ __align__(32) _Float16 Vt[64 * 64];       // [d][s] (transposed)
    __shared__ __align__(32) _Float16 Pl[8 * 16 * 64];   // per-wave P staging

    const int QB = T_ / 128;  // 16
    int nIdx = blockIdx.x / (H_ * QB);
    int rem  = blockIdx.x % (H_ * QB);
    int h    = rem / QB;
    int qb   = rem % QB;
    int qbase = qb * 128;
    int tid = threadIdx.x, lane = tid & 31, w = tid >> 5;
    int nlo = lane & 15, hi = lane >> 4;
    int qrow0 = qbase + w * 16;

    const _Float16* Qb = qh + ((size_t)(nIdx * H_ + h) * T_) * DH_;
    const _Float16* Kb = kh + ((size_t)(nIdx * H_ + h) * T_) * DH_;
    const _Float16* Vb = vh + ((size_t)(nIdx * H_ + h) * T_) * DH_;

    // Q strip as A-operands for both K-steps (d 0..31, 32..63), kept resident.
    v16h qa[2];
#pragma unroll
    for (int st = 0; st < 2; ++st) {
        const _Float16* qrow = Qb + (size_t)(qrow0 + nlo) * DH_;
        v8h c0 = *(const v8h*)(qrow + st * 32 + hi * 16);
        v8h c1 = *(const v8h*)(qrow + st * 32 + hi * 16 + 8);
        qa[st] = cat8(c0, c1);
    }

    v8f o[4];
#pragma unroll
    for (int nb = 0; nb < 4; ++nb) { v8f z = {}; o[nb] = z; }
    float mrow[8], lrow[8];
#pragma unroll
    for (int r = 0; r < 8; ++r) { mrow[r] = -__builtin_inff(); lrow[r] = 0.f; }

    int nblocks = qbase / 64 + 2;  // causal: key blocks covering rows <= qbase+127
    for (int jb = 0; jb < nblocks; ++jb) {
        int kbase = jb * 64;
        __syncthreads();
        {   // stage K tile [64][64]
            int s = tid >> 2, c = (tid & 3) * 16;
            const v8h* g = (const v8h*)(Kb + (size_t)(kbase + s) * DH_ + c);
            *(v8h*)(&Kl[s * 64 + c])     = g[0];
            *(v8h*)(&Kl[s * 64 + c + 8]) = g[1];
        }
        {   // stage V transposed: Vt[d][s]
            int s = tid >> 2, c = (tid & 3) * 16;
            const _Float16* g = Vb + (size_t)(kbase + s) * DH_ + c;
#pragma unroll
            for (int i = 0; i < 16; ++i) Vt[(c + i) * 64 + s] = g[i];
        }
        __syncthreads();

        // S = Q * K^T : 4 key sub-tiles of 16, 2 chained WMMAs (K=64 total)
        v8f sreg[4];
#pragma unroll
        for (int nb = 0; nb < 4; ++nb) {
            v8f a = {};
#pragma unroll
            for (int st = 0; st < 2; ++st) {
                v16h b = *(const v16h*)(&Kl[(nb * 16 + nlo) * 64 + st * 32 + hi * 16]);
                a = __builtin_amdgcn_wmma_f32_16x16x32_f16(
                        false, qa[st], false, b, (short)0, a, false, false);
            }
            int kg = kbase + nb * 16 + nlo;        // key index of this lane's column
#pragma unroll
            for (int r = 0; r < 8; ++r) {          // C layout: row = r + 8*hi
                int qg = qrow0 + r + 8 * hi;
                if (kg > qg) a[r] = -__builtin_inff();
            }
            sreg[nb] = a;
        }

        // online softmax: row max / rescale / rowsum (reduce over 16-lane halves)
        float mnew[8], alpha[8];
#pragma unroll
        for (int r = 0; r < 8; ++r) {
            float rm = sreg[0][r];
#pragma unroll
            for (int nb = 1; nb < 4; ++nb) rm = fmaxf(rm, sreg[nb][r]);
#pragma unroll
            for (int off = 8; off >= 1; off >>= 1)
                rm = fmaxf(rm, __shfl_xor(rm, off, 32));
            mnew[r]  = fmaxf(mrow[r], rm);
            alpha[r] = __expf(mrow[r] - mnew[r]);
        }
        float rs[8];
#pragma unroll
        for (int r = 0; r < 8; ++r) rs[r] = 0.f;
#pragma unroll
        for (int nb = 0; nb < 4; ++nb)
#pragma unroll
            for (int r = 0; r < 8; ++r) {
                float p = __expf(sreg[nb][r] - mnew[r]);
                sreg[nb][r] = p;
                rs[r] += p;
            }
#pragma unroll
        for (int r = 0; r < 8; ++r) {
#pragma unroll
            for (int off = 8; off >= 1; off >>= 1)
                rs[r] += __shfl_xor(rs[r], off, 32);
            lrow[r] = lrow[r] * alpha[r] + rs[r];
            mrow[r] = mnew[r];
        }
#pragma unroll
        for (int nb = 0; nb < 4; ++nb)
#pragma unroll
            for (int r = 0; r < 8; ++r) o[nb][r] *= alpha[r];

        // P: C-layout f32 -> A-layout f16 via per-wave LDS round-trip
        _Float16* Pw = &Pl[w * 16 * 64];
#pragma unroll
        for (int nb = 0; nb < 4; ++nb)
#pragma unroll
            for (int r = 0; r < 8; ++r)
                Pw[(r + 8 * hi) * 64 + nb * 16 + nlo] = (_Float16)sreg[nb][r];
        asm volatile("s_wait_dscnt 0" ::: "memory");
        v16h pa[2];
#pragma unroll
        for (int st = 0; st < 2; ++st) {
            v8h c0 = *(const v8h*)(&Pw[nlo * 64 + st * 32 + hi * 16]);
            v8h c1 = *(const v8h*)(&Pw[nlo * 64 + st * 32 + hi * 16 + 8]);
            pa[st] = cat8(c0, c1);
        }

        // O += P * V  (B-operand = Vt[d][s], contiguous s runs)
#pragma unroll
        for (int nb = 0; nb < 4; ++nb)
#pragma unroll
            for (int st = 0; st < 2; ++st) {
                v16h b = *(const v16h*)(&Vt[(nb * 16 + nlo) * 64 + st * 32 + hi * 16]);
                o[nb] = __builtin_amdgcn_wmma_f32_16x16x32_f16(
                            false, pa[st], false, b, (short)0, o[nb], false, false);
            }
    }

    // finalize: divide by l, write f16 [N*T, D] for the proj GEMM
#pragma unroll
    for (int nb = 0; nb < 4; ++nb)
#pragma unroll
        for (int r = 0; r < 8; ++r) {
            float val = o[nb][r] / lrow[r];
            int tq = qrow0 + r + 8 * hi;
            size_t off = ((size_t)nIdx * T_ + tq) * D_ + h * DH_ + nb * 16 + nlo;
            atth[off] = (_Float16)val;
        }
}

// ---------------------------------------------------------------------------
// Kernel 3: out = att @ proj_w.T + b  (fp32 output)
// proj_w.T means B-tiles Bt[n][k] = pw[n][k]: contiguous loads, no transpose.
// ---------------------------------------------------------------------------
__global__ __launch_bounds__(256) void proj_gemm(
        const _Float16* __restrict__ A,    // atth [4096][1024]
        const _Float16* __restrict__ Bw,   // pwh  [1024][1024] = pw[c][k]
        const float* __restrict__ bias,
        float* __restrict__ out) {
    __shared__ __align__(32) _Float16 As[128 * 32];
    __shared__ __align__(32) _Float16 Bs[64 * 32];
    const int Ntiles = D_ / 64;  // 16
    int mblock = (blockIdx.x / Ntiles) * 128;
    int cblock = (blockIdx.x % Ntiles) * 64;
    int tid = threadIdx.x, lane = tid & 31, w = tid >> 5;
    int wrow = w & 3, wcol = w >> 2;
    int nlo = lane & 15, hi = lane >> 4;

    v8f acc[2][2];
#pragma unroll
    for (int i = 0; i < 2; ++i)
#pragma unroll
        for (int j = 0; j < 2; ++j) { v8f z = {}; acc[i][j] = z; }

    for (int kb = 0; kb < D_; kb += 32) {
        __syncthreads();
        {
            int r = tid >> 1, c = (tid & 1) * 16;
            const v8h* g = (const v8h*)(A + (size_t)(mblock + r) * D_ + kb + c);
            *(v8h*)(&As[r * 32 + c])     = g[0];
            *(v8h*)(&As[r * 32 + c + 8]) = g[1];
        }
        if (tid < 128) {  // Bt[c][k] = pw[cblock+c][kb+k], contiguous
            int c = tid >> 1, k0 = (tid & 1) * 16;
            const v8h* g = (const v8h*)(Bw + (size_t)(cblock + c) * D_ + kb + k0);
            *(v8h*)(&Bs[c * 32 + k0])     = g[0];
            *(v8h*)(&Bs[c * 32 + k0 + 8]) = g[1];
        }
        __syncthreads();

        v16h aop[2], bop[2];
#pragma unroll
        for (int i = 0; i < 2; ++i) {
            int m = wrow * 32 + i * 16 + nlo;
            v8h c0 = *(const v8h*)(&As[m * 32 + hi * 16]);
            v8h c1 = *(const v8h*)(&As[m * 32 + hi * 16 + 8]);
            aop[i] = cat8(c0, c1);
        }
#pragma unroll
        for (int j = 0; j < 2; ++j) {
            int nn = wcol * 32 + j * 16 + nlo;
            bop[j] = *(const v16h*)(&Bs[nn * 32 + hi * 16]);
        }
#pragma unroll
        for (int i = 0; i < 2; ++i)
#pragma unroll
            for (int j = 0; j < 2; ++j)
                acc[i][j] = __builtin_amdgcn_wmma_f32_16x16x32_f16(
                    false, aop[i], false, bop[j], (short)0, acc[i][j], false, false);
    }

#pragma unroll
    for (int i = 0; i < 2; ++i)
#pragma unroll
        for (int j = 0; j < 2; ++j)
#pragma unroll
            for (int r = 0; r < 8; ++r) {
                int Rg = mblock + wrow * 32 + i * 16 + r + 8 * hi;
                int Cg = cblock + wcol * 32 + j * 16 + nlo;
                out[(size_t)Rg * D_ + Cg] = acc[i][j][r] + bias[Cg];
            }
}

// ---------------------------------------------------------------------------
extern "C" void kernel_launch(void* const* d_in, const int* in_sizes, int n_in,
                              void* d_out, int out_size, void* d_ws, size_t ws_size,
                              hipStream_t stream) {
    (void)in_sizes; (void)n_in; (void)out_size; (void)ws_size;
    const float* x  = (const float*)d_in[0];
    const float* W  = (const float*)d_in[1];
    const float* pw = (const float*)d_in[2];
    const float* pb = (const float*)d_in[3];
    float* out = (float*)d_out;

    char* ws = (char*)d_ws;
    const size_t MB = 1024 * 1024;
    _Float16* xh  = (_Float16*)(ws + 0 * MB);   // 8 MB
    _Float16* Wh  = (_Float16*)(ws + 8 * MB);   // 6 MB
    _Float16* pwh = (_Float16*)(ws + 14 * MB);  // 2 MB
    _Float16* qh  = (_Float16*)(ws + 16 * MB);  // 8 MB
    _Float16* kh  = (_Float16*)(ws + 24 * MB);  // 8 MB
    _Float16* vh  = (_Float16*)(ws + 32 * MB);  // 8 MB
    _Float16* ath = (_Float16*)(ws + 40 * MB);  // 8 MB

    int cx = N_ * T_ * D_;      // 4.19M
    int cw = D_ * TD3;          // 3.15M
    int cp = D_ * D_;           // 1.05M
    cvt_f32_to_f16<<<2048, 256, 0, stream>>>(x,  xh,  cx);
    cvt_f32_to_f16<<<2048, 256, 0, stream>>>(W,  Wh,  cw);
    cvt_f32_to_f16<<<1024, 256, 0, stream>>>(pw, pwh, cp);

    qkv_gemm<<<(N_ * T_ / 128) * (TD3 / 64), 256, 0, stream>>>(xh, Wh, qh, kh, vh);
    attn<<<N_ * H_ * (T_ / 128), 256, 0, stream>>>(qh, kh, vh, ath);
    proj_gemm<<<(N_ * T_ / 128) * (D_ / 64), 256, 0, stream>>>(ath, pwh, pb, out);
}
